// EncoderBlock_30485677867710
// MI455X (gfx1250) — matrix operand order repaired and checked by reading, not compiled, not run on hardware
//
#include <hip/hip_runtime.h>
#include <hip/hip_bf16.h>
#include <stdint.h>

// ---------------------------------------------------------------------------
// EncoderBlock for MI455X (gfx1250, wave32, WMMA).
// N=8192 rows, L=256 features, HID=128, H=2 heads, NLIN=2.
// All GEMMs: v_wmma_f32_16x16x32_f16. Flash-fused attention (no 256MB A).
// Fragment loads are all b128; B operands pre-transposed to row-per-column.
// Plain GEMMs use 32 output rows per wave so each B fragment feeds TWO
// WMMAs (1 ds_load_b128 per WMMA instead of 2 -> below LDS bank ceiling).
// Attention stages 64-key k/v^T tiles in LDS once per block (8x less L2).
// ---------------------------------------------------------------------------

#define NROWS 8192
#define LFEAT 256
#define HID   128
#define KV_TILE 64

typedef __attribute__((ext_vector_type(16))) _Float16 v16h;
typedef __attribute__((ext_vector_type(8)))  float    v8f;

union Frag16 { v16h v; _Float16 h[16]; uint32_t u[8]; uint4 q[2]; };

__device__ __forceinline__ v8f wmma_f16(v16h a, v16h b, v8f c) {
  // D = A(16x32 f16) * B(32x16 f16) + C(16x16 f32)
  return __builtin_amdgcn_wmma_f32_16x16x32_f16(
      false, a, false, b, (short)0, c, false, false);
}

// A-matrix 16x32 fragment from row-major src (ISA 7.12.2 layout):
// lane (g=l>>4, m=l&15) holds row m, K = {8g..8g+7, 16+8g..16+8g+7}.
__device__ __forceinline__ v16h frag_a(const _Float16* src, int ld, int row0, int k0) {
  const int lane = threadIdx.x & 31;
  const int g = lane >> 4, m = lane & 15;
  const _Float16* p = src + (size_t)(row0 + m) * ld + k0 + g * 8;
  Frag16 f;
  f.q[0] = *(const uint4*)(p);        // K = 8g + 0..7
  f.q[1] = *(const uint4*)(p + 16);   // K = 16 + 8g + 0..7
  return f.v;
}

// B-matrix 32x16 fragment where column n of B equals row (col0+n) of a
// row-major matrix: two 16-byte chunks per lane, same K order as frag_a.
__device__ __forceinline__ v16h frag_b_rows(const _Float16* src, int ld, int col0, int k0) {
  const int lane = threadIdx.x & 31;
  const int h2 = lane >> 4, n = lane & 15;
  const _Float16* p = src + (size_t)(col0 + n) * ld + k0 + h2 * 8;
  Frag16 f;
  f.q[0] = *(const uint4*)(p);        // K = 8*h2 + 0..7
  f.q[1] = *(const uint4*)(p + 16);   // K = 16 + 8*h2 + 0..7
  return f.v;
}

// ---------------------------------------------------------------------------
// conversion helpers
// ---------------------------------------------------------------------------
__global__ void f32_to_f16_kernel(const float* __restrict__ src,
                                  _Float16* __restrict__ dst, int n) {
  int i = blockIdx.x * blockDim.x + threadIdx.x;
  if (i < n) dst[i] = (_Float16)src[i];
}

// src [rows][cols] f32 (matrix blockIdx.y) -> dst [cols][rows] f16
__global__ void f32_to_f16_t_kernel(const float* __restrict__ src,
                                    _Float16* __restrict__ dst,
                                    int rows, int cols) {
  const float* s = src + (size_t)blockIdx.y * rows * cols;
  _Float16*   d = dst + (size_t)blockIdx.y * rows * cols;
  int i = blockIdx.x * blockDim.x + threadIdx.x;
  if (i < rows * cols) {
    int r = i / cols, c = i - r * cols;
    d[(size_t)c * rows + r] = (_Float16)s[i];
  }
}

// ---------------------------------------------------------------------------
// BatchNorm over batch dim (per feature column)
// ---------------------------------------------------------------------------
__global__ void bn_stats_kernel(const float* __restrict__ src,
                                float* __restrict__ mean,
                                float* __restrict__ rstd) {
  const int col = blockIdx.x;
  float s = 0.f, s2 = 0.f;
  for (int r = threadIdx.x; r < NROWS; r += 256) {
    float x = src[(size_t)r * LFEAT + col];
    s += x; s2 += x * x;
  }
  __shared__ float sh[256], sh2[256];
  sh[threadIdx.x] = s; sh2[threadIdx.x] = s2;
  __syncthreads();
  for (int off = 128; off > 0; off >>= 1) {
    if (threadIdx.x < off) {
      sh[threadIdx.x]  += sh[threadIdx.x + off];
      sh2[threadIdx.x] += sh2[threadIdx.x + off];
    }
    __syncthreads();
  }
  if (threadIdx.x == 0) {
    float m   = sh[0] * (1.f / NROWS);
    float var = sh2[0] * (1.f / NROWS) - m * m;
    mean[col] = m;
    rstd[col] = rsqrtf(var + 1e-5f);
  }
}

__global__ void bn_apply_kernel(const float* __restrict__ src,
                                const float* __restrict__ mean,
                                const float* __restrict__ rstd,
                                float* __restrict__ out32,
                                _Float16* __restrict__ out16) {
  int i = blockIdx.x * 256 + threadIdx.x;
  int col = i & (LFEAT - 1);
  float v = (src[i] - mean[col]) * rstd[col];
  if (out32) out32[i] = v;
  out16[i] = (_Float16)v;
}

// ---------------------------------------------------------------------------
// Projections: q/k row-major [N][HID]; v_down written TRANSPOSED [HID][N].
// Weights arrive pre-transposed [HID][L], staged in 64KB LDS.
// 32 rows per wave: each B fragment feeds two WMMAs.
// blockIdx.y = matrix index 0..5 ({Q,K,Vd} x {head0,head1}).
// ---------------------------------------------------------------------------
__global__ __launch_bounds__(256)
void proj_gemm_kernel(const _Float16* __restrict__ h16,
                      const _Float16* __restrict__ Qt,
                      const _Float16* __restrict__ Kt,
                      const _Float16* __restrict__ Vdt,
                      _Float16* __restrict__ q16,
                      _Float16* __restrict__ k16,
                      _Float16* __restrict__ vd16t) {
  __shared__ _Float16 Bs[HID * LFEAT];  // 64 KB weight^T tile [128][256]
  const int mat  = blockIdx.y;
  const int head = mat & 1;
  const _Float16* wsrc = ((mat < 2) ? Qt : (mat < 4) ? Kt : Vdt)
                         + (size_t)head * HID * LFEAT;

  const uint4* gs = (const uint4*)wsrc;
  uint4* ls = (uint4*)Bs;
  for (int i = threadIdx.x; i < (HID * LFEAT) / 8; i += 256) ls[i] = gs[i];
  __syncthreads();

  const int wave = threadIdx.x >> 5;
  const int row0 = blockIdx.x * 256 + wave * 32;   // 32 rows per wave
  v8f acc[16];
#pragma unroll
  for (int c = 0; c < 16; ++c) acc[c] = {};

  for (int kk = 0; kk < LFEAT; kk += 32) {
    v16h a0 = frag_a(h16, LFEAT, row0,      kk);
    v16h a1 = frag_a(h16, LFEAT, row0 + 16, kk);
#pragma unroll
    for (int c = 0; c < 8; ++c) {
      v16h bf = frag_b_rows(Bs, LFEAT, c * 16, kk);
      acc[c]     = wmma_f16(a0, bf, acc[c]);       // B reused by both rows
      acc[8 + c] = wmma_f16(a1, bf, acc[8 + c]);
    }
  }

  const int lane = threadIdx.x & 31;
  const int rOff = (lane >> 4) * 8, n = lane & 15;
  if (mat < 4) {
    _Float16* dst = ((mat < 2) ? q16 : k16) + (size_t)head * NROWS * HID;
#pragma unroll
    for (int half = 0; half < 2; ++half)
#pragma unroll
      for (int c = 0; c < 8; ++c)
#pragma unroll
        for (int vv = 0; vv < 8; ++vv)
          dst[(size_t)(row0 + half * 16 + rOff + vv) * HID + c * 16 + n] =
              (_Float16)acc[half * 8 + c][vv];
  } else {
    _Float16* dstT = vd16t + (size_t)head * NROWS * HID;
#pragma unroll
    for (int half = 0; half < 2; ++half)
#pragma unroll
      for (int c = 0; c < 8; ++c) {  // lane owns feature d=c*16+n, 8 rows->b128
        Frag16 pk;
#pragma unroll
        for (int vv = 0; vv < 8; ++vv) pk.h[vv] = (_Float16)acc[half * 8 + c][vv];
        *(uint4*)(dstT + (size_t)(c * 16 + n) * NROWS + row0 + half * 16 + rOff) =
            pk.q[0];
      }
  }
}

// ---------------------------------------------------------------------------
// Flash attention per head: O = softmax(q k^T) @ vd (vd transposed [HID][N]).
// Block = 128 query rows (8 waves x 16 rows). 64-key k/v^T tiles are staged
// in LDS cooperatively (one global read per block, consumed by 8 waves).
// ---------------------------------------------------------------------------
__global__ __launch_bounds__(256)
void flash_attn_kernel(const _Float16* __restrict__ q16,
                       const _Float16* __restrict__ k16,
                       const _Float16* __restrict__ vd16t,
                       _Float16* __restrict__ o16) {
  const size_t hoff = (size_t)blockIdx.y * NROWS * HID;
  const _Float16* q  = q16   + hoff;
  const _Float16* k  = k16   + hoff;
  const _Float16* vt = vd16t + hoff;   // [HID][NROWS]
  _Float16* o = o16 + hoff;

  __shared__ _Float16 Ks[KV_TILE * HID];    // [64][128] 16 KB
  __shared__ _Float16 Vts[HID * KV_TILE];   // [128][64] 16 KB
  __shared__ _Float16 Ps[8][16 * 32];       // per-wave prob tiles, 8 KB

  const int wave = threadIdx.x >> 5;
  const int row0 = blockIdx.x * 128 + wave * 16;
  const int lane = threadIdx.x & 31;
  const int g = lane >> 4, n = lane & 15;

  v8f acc[8];
#pragma unroll
  for (int c = 0; c < 8; ++c) acc[c] = {};
  float mrow[8], lrow[8];
#pragma unroll
  for (int vv = 0; vv < 8; ++vv) { mrow[vv] = -3.0e38f; lrow[vv] = 0.f; }

  v16h aq[4];  // q fragments (K=128), hoisted for the whole kv stream
#pragma unroll
  for (int kkk = 0; kkk < 4; ++kkk) aq[kkk] = frag_a(q, HID, row0, kkk * 32);

  for (int kv = 0; kv < NROWS; kv += KV_TILE) {
    __syncthreads();  // previous tile fully consumed by all waves
    {                 // cooperative staging: 2048 x b128 across 256 threads
      const uint4* gk = (const uint4*)(k + (size_t)kv * HID);
      uint4* lk = (uint4*)Ks;
#pragma unroll
      for (int t = 0; t < 4; ++t)
        lk[threadIdx.x + 256 * t] = gk[threadIdx.x + 256 * t];
      uint4* lv = (uint4*)Vts;
#pragma unroll
      for (int t = 0; t < 4; ++t) {
        int idx = threadIdx.x + 256 * t;       // 0..1023
        int d = idx >> 3, j = idx & 7;         // feature row d, 8-half chunk j
        lv[idx] = *(const uint4*)(vt + (size_t)d * NROWS + kv + j * 8);
      }
    }
    __syncthreads();
    if (kv + KV_TILE < NROWS) {
      __builtin_prefetch(k + (size_t)(kv + KV_TILE) * HID, 0, 1);
      __builtin_prefetch(vt + kv + KV_TILE, 0, 1);
    }

#pragma unroll
    for (int sub = 0; sub < 2; ++sub) {
      const int kvl = sub * 32;
      // ---- S = q k^T over 32 keys ----
      v16h bk[8];
#pragma unroll
      for (int kkk = 0; kkk < 4; ++kkk) {
        bk[kkk]     = frag_b_rows(Ks, HID, kvl,      kkk * 32);
        bk[4 + kkk] = frag_b_rows(Ks, HID, kvl + 16, kkk * 32);
      }
      v8f s0 = {}, s1 = {};
#pragma unroll
      for (int kkk = 0; kkk < 4; ++kkk) {
        s0 = wmma_f16(aq[kkk], bk[kkk],     s0);
        s1 = wmma_f16(aq[kkk], bk[4 + kkk], s1);
      }
      // ---- online softmax (reference applies no 1/sqrt(d) scale) ----
#pragma unroll
      for (int vv = 0; vv < 8; ++vv) {
        float bm = fmaxf(s0[vv], s1[vv]);
        bm = fmaxf(bm, __shfl_xor(bm, 1));
        bm = fmaxf(bm, __shfl_xor(bm, 2));
        bm = fmaxf(bm, __shfl_xor(bm, 4));
        bm = fmaxf(bm, __shfl_xor(bm, 8));   // row max across 16 lanes
        float mnew = fmaxf(mrow[vv], bm);
        float cf = __expf(mrow[vv] - mnew);
        float p0 = __expf(s0[vv] - mnew);
        float p1 = __expf(s1[vv] - mnew);
        float ps = p0 + p1;
        ps += __shfl_xor(ps, 1); ps += __shfl_xor(ps, 2);
        ps += __shfl_xor(ps, 4); ps += __shfl_xor(ps, 8);
        lrow[vv] = lrow[vv] * cf + ps;
        mrow[vv] = mnew;
#pragma unroll
        for (int c = 0; c < 8; ++c) acc[c][vv] *= cf;  // rescale running O
        Ps[wave][(vv + g * 8) * 32 + n]      = (_Float16)p0;
        Ps[wave][(vv + g * 8) * 32 + 16 + n] = (_Float16)p1;
      }
      // ---- O += P(16x32) @ V(32x128) ----
      v16h ap = frag_a(&Ps[wave][0], 32, 0, 0);
      v16h bv[8];
#pragma unroll
      for (int c = 0; c < 8; ++c)
        bv[c] = frag_b_rows(Vts, KV_TILE, c * 16, kvl);
#pragma unroll
      for (int c = 0; c < 8; ++c) acc[c] = wmma_f16(ap, bv[c], acc[c]);
    }
  }

#pragma unroll
  for (int vv = 0; vv < 8; ++vv) {
    float inv = 1.0f / lrow[vv];
#pragma unroll
    for (int c = 0; c < 8; ++c)
      o[(size_t)(row0 + g * 8 + vv) * HID + c * 16 + n] =
          (_Float16)(acc[c][vv] * inv);
  }
}

// ---------------------------------------------------------------------------
// dE = sum_heads O_h @ V_up_h (V_up pre-transposed [L][HID]),
// added in place onto h32. 32 rows per wave (B frag feeds two WMMAs).
// ---------------------------------------------------------------------------
__global__ __launch_bounds__(256)
void devalue_gemm_kernel(const _Float16* __restrict__ o16,
                         const _Float16* __restrict__ vupT,
                         float* __restrict__ h32) {
  const int wave = threadIdx.x >> 5;
  const int row0 = blockIdx.x * 256 + wave * 32;
  const int col0 = blockIdx.y * 128;
  v8f acc[16];
#pragma unroll
  for (int c = 0; c < 16; ++c) acc[c] = {};

  for (int head = 0; head < 2; ++head) {
    const _Float16* O  = o16  + (size_t)head * NROWS * HID;
    const _Float16* Vu = vupT + (size_t)head * LFEAT * HID;  // [256][128]
#pragma unroll
    for (int kk = 0; kk < HID; kk += 32) {
      v16h a0 = frag_a(O, HID, row0,      kk);
      v16h a1 = frag_a(O, HID, row0 + 16, kk);
#pragma unroll
      for (int c = 0; c < 8; ++c) {
        v16h bf = frag_b_rows(Vu, HID, col0 + c * 16, kk);
        acc[c]     = wmma_f16(a0, bf, acc[c]);
        acc[8 + c] = wmma_f16(a1, bf, acc[8 + c]);
      }
    }
  }
  const int lane = threadIdx.x & 31;
  const int rOff = (lane >> 4) * 8, n = lane & 15;
#pragma unroll
  for (int half = 0; half < 2; ++half)
#pragma unroll
    for (int c = 0; c < 8; ++c)
#pragma unroll
      for (int vv = 0; vv < 8; ++vv) {
        size_t idx = (size_t)(row0 + half * 16 + rOff + vv) * LFEAT
                     + col0 + c * 16 + n;
        h32[idx] += acc[half * 8 + c][vv];  // residual h + dE, in place
      }
}

// ---------------------------------------------------------------------------
// h = relu(h @ W^T + b). W rows for this col-block staged in 64KB LDS.
// 32 rows per wave. Last layer writes f32 to d_out.
// ---------------------------------------------------------------------------
__global__ __launch_bounds__(256)
void linear_relu_kernel(const _Float16* __restrict__ A,
                        const _Float16* __restrict__ Wf,  // [256][256] row-major
                        const float* __restrict__ bias,   // [256]
                        _Float16* __restrict__ out16,
                        float* __restrict__ out32) {
  __shared__ _Float16 Ws[128 * LFEAT];  // 64 KB: W rows col0..col0+127
  const int wave = threadIdx.x >> 5;
  const int row0 = blockIdx.x * 256 + wave * 32;
  const int col0 = blockIdx.y * 128;

  const uint4* gs = (const uint4*)(Wf + (size_t)col0 * LFEAT);
  uint4* ls = (uint4*)Ws;
  for (int i = threadIdx.x; i < (128 * LFEAT) / 8; i += 256) ls[i] = gs[i];
  __syncthreads();

  v8f acc[16];
#pragma unroll
  for (int c = 0; c < 16; ++c) acc[c] = {};

  for (int kk = 0; kk < LFEAT; kk += 32) {
    v16h a0 = frag_a(A, LFEAT, row0,      kk);
    v16h a1 = frag_a(A, LFEAT, row0 + 16, kk);
#pragma unroll
    for (int c = 0; c < 8; ++c) {
      v16h bf = frag_b_rows(Ws, LFEAT, c * 16, kk);
      acc[c]     = wmma_f16(a0, bf, acc[c]);
      acc[8 + c] = wmma_f16(a1, bf, acc[8 + c]);
    }
  }
  const int lane = threadIdx.x & 31;
  const int rOff = (lane >> 4) * 8, n = lane & 15;
#pragma unroll
  for (int half = 0; half < 2; ++half)
#pragma unroll
    for (int c = 0; c < 8; ++c)
#pragma unroll
      for (int vv = 0; vv < 8; ++vv) {
        int col = col0 + c * 16 + n;
        float val = fmaxf(acc[half * 8 + c][vv] + bias[col], 0.f);
        size_t idx = (size_t)(row0 + half * 16 + rOff + vv) * LFEAT + col;
        if (out32) out32[idx] = val;
        else       out16[idx] = (_Float16)val;
      }
}

// ---------------------------------------------------------------------------
// host-side orchestration
// ---------------------------------------------------------------------------
extern "C" void kernel_launch(void* const* d_in, const int* in_sizes, int n_in,
                              void* d_out, int out_size, void* d_ws, size_t ws_size,
                              hipStream_t stream) {
  const float* x   = (const float*)d_in[0];
  const float* Qw  = (const float*)d_in[1];   // [2][256][128]
  const float* Kw  = (const float*)d_in[2];   // [2][256][128]
  const float* Vdw = (const float*)d_in[3];   // [2][256][128]
  const float* Vuw = (const float*)d_in[4];   // [2][128][256]
  const float* Ww  = (const float*)d_in[5];   // [2][256][256]
  const float* bw  = (const float*)d_in[6];   // [2][256]
  float* out = (float*)d_out;

  char* ws = (char*)d_ws;
  size_t off = 0;
  auto carve = [&](size_t bytes) -> void* {
    off = (off + 255) & ~(size_t)255;
    void* p = ws + off;
    off += bytes;
    return p;
  };
  float*    mean1 = (float*)carve(LFEAT * 4);
  float*    rstd1 = (float*)carve(LFEAT * 4);
  float*    mean2 = (float*)carve(LFEAT * 4);
  float*    rstd2 = (float*)carve(LFEAT * 4);
  _Float16* h16   = (_Float16*)carve((size_t)NROWS * LFEAT * 2);  // 4 MB
  float*    h32   = (float*)   carve((size_t)NROWS * LFEAT * 4);  // 8 MB
  _Float16* q16   = (_Float16*)carve((size_t)2 * NROWS * HID * 2);
  _Float16* k16   = (_Float16*)carve((size_t)2 * NROWS * HID * 2);
  _Float16* vd16t = (_Float16*)carve((size_t)2 * NROWS * HID * 2); // [H][HID][N]
  _Float16* o16   = (_Float16*)carve((size_t)2 * NROWS * HID * 2);
  _Float16* h2    = (_Float16*)carve((size_t)NROWS * LFEAT * 2);
  _Float16* h3    = (_Float16*)carve((size_t)NROWS * LFEAT * 2);
  _Float16* qt    = (_Float16*)carve((size_t)2 * LFEAT * HID * 2); // [H][HID][L]
  _Float16* kt    = (_Float16*)carve((size_t)2 * LFEAT * HID * 2);
  _Float16* vdt   = (_Float16*)carve((size_t)2 * LFEAT * HID * 2);
  _Float16* vut   = (_Float16*)carve((size_t)2 * HID * LFEAT * 2); // [H][L][HID]
  _Float16* wf    = (_Float16*)carve((size_t)2 * LFEAT * LFEAT * 2);

  const int nmat = LFEAT * HID;          // 32768 per head
  const int nw   = 2 * LFEAT * LFEAT;    // 131072
  dim3 cgrid((nmat + 255) / 256, 2);
  f32_to_f16_t_kernel<<<cgrid, 256, 0, stream>>>(Qw,  qt,  LFEAT, HID);
  f32_to_f16_t_kernel<<<cgrid, 256, 0, stream>>>(Kw,  kt,  LFEAT, HID);
  f32_to_f16_t_kernel<<<cgrid, 256, 0, stream>>>(Vdw, vdt, LFEAT, HID);
  f32_to_f16_t_kernel<<<cgrid, 256, 0, stream>>>(Vuw, vut, HID, LFEAT);
  f32_to_f16_kernel<<<(nw + 255) / 256, 256, 0, stream>>>(Ww, wf, nw);

  // BN1
  bn_stats_kernel<<<LFEAT, 256, 0, stream>>>(x, mean1, rstd1);
  bn_apply_kernel<<<NROWS, 256, 0, stream>>>(x, mean1, rstd1, h32, h16);

  // q/k/v_down projections (v_down emitted transposed); 256 rows per block
  proj_gemm_kernel<<<dim3(NROWS / 256, 6), 256, 0, stream>>>(
      h16, qt, kt, vdt, q16, k16, vd16t);

  // flash attention per head
  flash_attn_kernel<<<dim3(NROWS / 128, 2), 256, 0, stream>>>(q16, k16, vd16t, o16);

  // h32 += sum_h O_h @ V_up_h
  devalue_gemm_kernel<<<dim3(NROWS / 256, LFEAT / 128), 256, 0, stream>>>(o16, vut, h32);

  // BN2 + two Linear/ReLU layers
  bn_stats_kernel<<<LFEAT, 256, 0, stream>>>(h32, mean2, rstd2);
  bn_apply_kernel<<<NROWS, 256, 0, stream>>>(h32, mean2, rstd2, nullptr, h2);

  linear_relu_kernel<<<dim3(NROWS / 256, LFEAT / 128), 256, 0, stream>>>(
      h2, wf, bw, h3, nullptr);
  linear_relu_kernel<<<dim3(NROWS / 256, LFEAT / 128), 256, 0, stream>>>(
      h3, wf + (size_t)LFEAT * LFEAT, bw + LFEAT, nullptr, out);
}
